// Transformer_78469052498176
// MI455X (gfx1250) — matrix operand order repaired and checked by reading, not compiled, not run on hardware
//
#include <hip/hip_runtime.h>

typedef __attribute__((ext_vector_type(16))) _Float16 v16h;
typedef __attribute__((ext_vector_type(8)))  _Float16 v8h;
typedef __attribute__((ext_vector_type(4)))  _Float16 v4h;
typedef __attribute__((ext_vector_type(8)))  float    v8f;

#define TT 512
#define DD 512
#define HH 8
#define HDH 64
#define BBATCH 16
#define VV 50304
#define PAD_ID 50259

static __device__ inline v16h cat8(v8h lo, v8h hi) {
  v16h r;
#pragma unroll
  for (int i = 0; i < 8; ++i) { r[i] = lo[i]; r[i + 8] = hi[i]; }
  return r;
}

static __device__ inline v4h cvt4(float x, float y, float z, float w) {
  v4h r;
  r[0] = (_Float16)x; r[1] = (_Float16)y; r[2] = (_Float16)z; r[3] = (_Float16)w;
  return r;
}

static __device__ inline v8f wmma16(v16h a, v16h b, v8f c) {
  return __builtin_amdgcn_wmma_f32_16x16x32_f16(false, a, false, b, (short)0, c, false, false);
}

// ---------------------------------------------------------------------------
// Embedding: out[b,t,:] = emb[id]*sqrt(D) + pos_enc(t,:)   (one float4/thread)
// ---------------------------------------------------------------------------
__global__ __launch_bounds__(128) void embed_kernel(const int* __restrict__ ids,
                                                    const float* __restrict__ emb,
                                                    float* __restrict__ out) {
  int bt = blockIdx.x;
  int t = bt % TT;
  int id = ids[bt];
  int d0 = threadIdx.x * 4;
  float4 tok = *(const float4*)(emb + (size_t)id * DD + d0);
  float r[4] = {tok.x, tok.y, tok.z, tok.w};
  float4 o;
  float* op = (float*)&o;
#pragma unroll
  for (int j = 0; j < 4; ++j) {
    int d = d0 + j;
    int i2 = d >> 1;
    float freq = __expf(-(float)(2 * i2) * (9.210340371976184f / 512.0f)); // 10000^(-2i/D)
    float ang = (float)t * freq;
    float pe = (d & 1) ? __cosf(ang) : __sinf(ang);
    op[j] = r[j] * 22.62741699796952f + pe; // sqrt(512)
  }
  *(float4*)(out + (size_t)bt * DD + d0) = o;
}

// ---------------------------------------------------------------------------
// LayerNorm over rows of 512 (one float4/thread)
// ---------------------------------------------------------------------------
__global__ __launch_bounds__(128) void ln_kernel(const float* __restrict__ in,
                                                 const float* __restrict__ g,
                                                 const float* __restrict__ bb,
                                                 float* __restrict__ out) {
  __shared__ float red[128];
  int row = blockIdx.x;
  int t = threadIdx.x;
  float4 xv = *(const float4*)(in + (size_t)row * DD + t * 4);
  float x[4] = {xv.x, xv.y, xv.z, xv.w};
  float s = x[0] + x[1] + x[2] + x[3];
  red[t] = s;
  __syncthreads();
  for (int off = 64; off > 0; off >>= 1) {
    if (t < off) red[t] += red[t + off];
    __syncthreads();
  }
  float mean = red[0] * (1.0f / 512.0f);
  __syncthreads();
  float s2 = 0.0f;
#pragma unroll
  for (int i = 0; i < 4; ++i) { float d = x[i] - mean; s2 += d * d; }
  red[t] = s2;
  __syncthreads();
  for (int off = 64; off > 0; off >>= 1) {
    if (t < off) red[t] += red[t + off];
    __syncthreads();
  }
  float rstd = rsqrtf(red[0] * (1.0f / 512.0f) + 1e-5f);
  float4 gv = *(const float4*)(g + t * 4);
  float4 bv = *(const float4*)(bb + t * 4);
  float4 o;
  o.x = (x[0] - mean) * rstd * gv.x + bv.x;
  o.y = (x[1] - mean) * rstd * gv.y + bv.y;
  o.z = (x[2] - mean) * rstd * gv.z + bv.z;
  o.w = (x[3] - mean) * rstd * gv.w + bv.w;
  *(float4*)(out + (size_t)row * DD + t * 4) = o;
}

// ---------------------------------------------------------------------------
// WMMA GEMM: C[M,N] = A[M,K](lda) * W[K,N] + bias (+residual) (relu?)
// block tile 64x64, BK=64, 4 waves, each wave 32x32 (2x2 WMMA tiles)
// ---------------------------------------------------------------------------
#define AS_STRIDE 72
#define BS_STRIDE 72

__global__ __launch_bounds__(128) void gemm_kernel(const float* __restrict__ A, int lda,
                                                   const float* __restrict__ W,
                                                   const float* __restrict__ bias,
                                                   const float* __restrict__ residual,
                                                   float* __restrict__ C,
                                                   int M, int N, int K, int relu) {
  __shared__ _Float16 As[64 * AS_STRIDE];
  __shared__ _Float16 Bs[64 * BS_STRIDE];

  int tid = threadIdx.x;
  int lane = tid & 31;
  int w = tid >> 5;
  int wm = w & 1;
  int wn = w >> 1;
  int row0 = blockIdx.y * 64;
  int col0 = blockIdx.x * 64;

  v8f zero8 = {0.f, 0.f, 0.f, 0.f, 0.f, 0.f, 0.f, 0.f};
  v8f acc[2][2];
  acc[0][0] = zero8; acc[0][1] = zero8; acc[1][0] = zero8; acc[1][1] = zero8;

  int l16 = lane & 15;
  int khalf = lane >> 4;
  bool full = (row0 + 64) <= M;   // uniform across block

  for (int k0 = 0; k0 < K; k0 += 64) {
    // ---- stage A tile 64x64 (branch-free: clamp row, zero-select) ----
    float4 av[8];
#pragma unroll
    for (int i = 0; i < 8; ++i) {
      int li = i * 128 + tid;      // 0..1023 float4 groups
      int r = li >> 4;             // row 0..63
      int c4 = li & 15;
      int gr = row0 + r;
      int grc = gr < M ? gr : 0;
      av[i] = *(const float4*)(A + (size_t)grc * lda + k0 + c4 * 4);
    }
    if (full) {
#pragma unroll
      for (int i = 0; i < 8; ++i) {
        int li = i * 128 + tid;
        int r = li >> 4, c4 = li & 15;
        *(v4h*)&As[r * AS_STRIDE + c4 * 4] = cvt4(av[i].x, av[i].y, av[i].z, av[i].w);
      }
    } else {
#pragma unroll
      for (int i = 0; i < 8; ++i) {
        int li = i * 128 + tid;
        int r = li >> 4, c4 = li & 15;
        float z = (row0 + r) < M ? 1.0f : 0.0f;
        *(v4h*)&As[r * AS_STRIDE + c4 * 4] =
            cvt4(av[i].x * z, av[i].y * z, av[i].z * z, av[i].w * z);
      }
    }
    // ---- stage W tile 64x64 transposed -> Bs[n][k] (4 coalesced b32 / group)
#pragma unroll
    for (int gq = 0; gq < 8; ++gq) {
      int li = gq * 128 + tid;     // 0..1023
      int n = li & 63;
      int k4 = li >> 6;            // 0..15
      const float* wp = W + (size_t)(k0 + k4 * 4) * N + col0 + n;
      float f0 = wp[0];
      float f1 = wp[(size_t)N];
      float f2 = wp[(size_t)2 * N];
      float f3 = wp[(size_t)3 * N];
      *(v4h*)&Bs[n * BS_STRIDE + k4 * 4] = cvt4(f0, f1, f2, f3);
    }
    __syncthreads();

#pragma unroll
    for (int d0 = 0; d0 < 64; d0 += 32) {
      v16h af[2], bf[2];
#pragma unroll
      for (int mi = 0; mi < 2; ++mi) {
        const _Float16* ap = &As[(wm * 32 + mi * 16 + l16) * AS_STRIDE + d0];
        af[mi] = cat8(*(const v8h*)(ap + khalf * 8), *(const v8h*)(ap + 16 + khalf * 8));
      }
#pragma unroll
      for (int ni = 0; ni < 2; ++ni) {
        const _Float16* bp = &Bs[(wn * 32 + ni * 16 + l16) * BS_STRIDE + d0];
        bf[ni] = cat8(*(const v8h*)(bp + khalf * 8), *(const v8h*)(bp + 16 + khalf * 8));
      }
#pragma unroll
      for (int mi = 0; mi < 2; ++mi)
#pragma unroll
        for (int ni = 0; ni < 2; ++ni)
          acc[mi][ni] = wmma16(af[mi], bf[ni], acc[mi][ni]);
    }
    __syncthreads();
  }

  int lhi = lane >> 4;
  if (full) {
    // uniform fast path: no per-lane guards
#pragma unroll
    for (int mi = 0; mi < 2; ++mi)
#pragma unroll
      for (int ni = 0; ni < 2; ++ni) {
        int col = col0 + wn * 32 + ni * 16 + l16;
        float bc = bias[col];
        int rbase = row0 + wm * 32 + mi * 16 + lhi * 8;
        float v[8];
#pragma unroll
        for (int r = 0; r < 8; ++r) {
          v[r] = acc[mi][ni][r] + bc;
          if (relu) v[r] = fmaxf(v[r], 0.0f);
        }
        if (residual) {
#pragma unroll
          for (int r = 0; r < 8; ++r)
            v[r] += residual[(size_t)(rbase + r) * N + col];
        }
#pragma unroll
        for (int r = 0; r < 8; ++r)
          C[(size_t)(rbase + r) * N + col] = v[r];
      }
  } else {
#pragma unroll
    for (int mi = 0; mi < 2; ++mi)
#pragma unroll
      for (int ni = 0; ni < 2; ++ni)
#pragma unroll
        for (int r = 0; r < 8; ++r) {
          int row = row0 + wm * 32 + mi * 16 + lhi * 8 + r;
          int col = col0 + wn * 32 + ni * 16 + l16;
          if (row < M) {
            float v = acc[mi][ni][r] + bias[col];
            if (relu) v = fmaxf(v, 0.0f);
            if (residual) v += residual[(size_t)row * N + col];
            C[(size_t)row * N + col] = v;
          }
        }
  }
}

// ---------------------------------------------------------------------------
// Fused attention: one block per (b, h, 16-query tile).
// scores (WMMA) -> masked softmax in LDS -> probs @ V (WMMA)
// dynamic LDS: qs 16x80 f16 | kv 128x80 f16 (ks) / 64x136 f16 (vs) |
//              ss 16x512 f32 | red 16x8 f32 | padk 512 u8  = 56832 bytes
// ---------------------------------------------------------------------------
#define ATTN_SMEM 56832

__global__ __launch_bounds__(128) void attn_kernel(const float* __restrict__ q,
                                                   const float* __restrict__ kmat,
                                                   const float* __restrict__ vmat,
                                                   float* __restrict__ out,
                                                   const int* __restrict__ ids_q,
                                                   const int* __restrict__ ids_k,
                                                   int causal) {
  extern __shared__ char smem[];
  _Float16* qs = (_Float16*)smem;                    // 16*80 halfs
  _Float16* kv = (_Float16*)(smem + 2560);           // 128*80 halfs (ks) / 64*136 (vs)
  float* ss = (float*)(smem + 23040);                // 16*512 floats
  float* red = (float*)(smem + 55808);               // 16*8 floats
  unsigned char* padk = (unsigned char*)(smem + 56320); // 512 flags

  int b = blockIdx.z;
  int h = blockIdx.y;
  int q0 = blockIdx.x * 16;
  int tid = threadIdx.x;
  int lane = tid & 31;
  int w = tid >> 5;
  int l16 = lane & 15;
  int khalf = lane >> 4;

  // stage Q tile (16 x 64) as f16, and key-pad flags
#pragma unroll
  for (int i = 0; i < 2; ++i) {
    int li = i * 128 + tid;
    int r = li >> 4, c4 = li & 15;
    float4 f = *(const float4*)(q + ((size_t)(b * TT + q0 + r)) * DD + h * HDH + c4 * 4);
    *(v4h*)&qs[r * 80 + c4 * 4] = cvt4(f.x, f.y, f.z, f.w);
  }
#pragma unroll
  for (int i = 0; i < 4; ++i) {
    int idx = i * 128 + tid;
    padk[idx] = (ids_k[b * TT + idx] != PAD_ID) ? 1 : 0;
  }

  v8f zero8 = {0.f, 0.f, 0.f, 0.f, 0.f, 0.f, 0.f, 0.f};

  // ---- phase 1: scores ----
  for (int c = 0; c < 4; ++c) {
    // stage 128 keys x 64 as f16 (row = key, col = d)
#pragma unroll
    for (int gq = 0; gq < 4; ++gq) {
      float4 f[4];
#pragma unroll
      for (int u = 0; u < 4; ++u) {
        int li = (gq * 4 + u) * 128 + tid;
        int kk = li >> 4, c4 = li & 15;
        f[u] = *(const float4*)(kmat + ((size_t)(b * TT + c * 128 + kk)) * DD + h * HDH + c4 * 4);
      }
#pragma unroll
      for (int u = 0; u < 4; ++u) {
        int li = (gq * 4 + u) * 128 + tid;
        int kk = li >> 4, c4 = li & 15;
        *(v4h*)&kv[kk * 80 + c4 * 4] = cvt4(f[u].x, f[u].y, f[u].z, f[u].w);
      }
    }
    __syncthreads();

    for (int j = 0; j < 2; ++j) {
      int kt = w * 2 + j;
      v8f acc = zero8;
#pragma unroll
      for (int d0 = 0; d0 < 64; d0 += 32) {
        const _Float16* ap = &qs[l16 * 80 + d0];
        v16h af = cat8(*(const v8h*)(ap + khalf * 8), *(const v8h*)(ap + 16 + khalf * 8));
        const _Float16* bp = &kv[(kt * 16 + l16) * 80 + d0];
        v16h bf = cat8(*(const v8h*)(bp + khalf * 8), *(const v8h*)(bp + 16 + khalf * 8));
        acc = wmma16(af, bf, acc);
      }
#pragma unroll
      for (int r = 0; r < 8; ++r) {
        int m = khalf * 8 + r;
        ss[m * 512 + c * 128 + kt * 16 + l16] = acc[r] * 0.125f; // 1/sqrt(64)
      }
    }
    __syncthreads();
  }

  // ---- phase 2: masked softmax over 512 keys ----
  {
    int r = tid & 15;
    int seg = tid >> 4; // 0..7, 64 cols each
    int qg = q0 + r;
    bool padq = (ids_q[b * TT + qg] != PAD_ID);

    float mx = -3.0e38f;
    for (int cc = 0; cc < 64; ++cc) {
      int col = seg * 64 + cc;
      bool valid = padq && padk[col] && (!causal || col <= qg);
      float vv = valid ? ss[r * 512 + col] : -1.0e9f;
      mx = fmaxf(mx, vv);
    }
    red[r * 8 + seg] = mx;
    __syncthreads();
    float rowmax = red[r * 8];
#pragma unroll
    for (int i = 1; i < 8; ++i) rowmax = fmaxf(rowmax, red[r * 8 + i]);
    __syncthreads();

    float sum = 0.0f;
    for (int cc = 0; cc < 64; ++cc) {
      int col = seg * 64 + cc;
      bool valid = padq && padk[col] && (!causal || col <= qg);
      float vv = valid ? ss[r * 512 + col] : -1.0e9f;
      float p = __expf(vv - rowmax);
      ss[r * 512 + col] = p;
      sum += p;
    }
    red[r * 8 + seg] = sum;
    __syncthreads();
    float tot = 0.0f;
#pragma unroll
    for (int i = 0; i < 8; ++i) tot += red[r * 8 + i];
    float inv = 1.0f / tot;
    __syncthreads();
    for (int cc = 0; cc < 64; ++cc) {
      int col = seg * 64 + cc;
      ss[r * 512 + col] *= inv;
    }
    __syncthreads();
  }

  // ---- phase 3: probs @ V ----
  int n0 = w * 16;
  v8f oacc = zero8;
  for (int c = 0; c < 4; ++c) {
    // stage V chunk transposed: vs[d][key_local], stride 136
#pragma unroll
    for (int gq = 0; gq < 4; ++gq) {
      float f[4][4];
#pragma unroll
      for (int u = 0; u < 4; ++u) {
        int li = (gq * 4 + u) * 128 + tid;
        int d = li & 63;
        int k4 = li >> 6; // 0..31
        const float* vp = vmat + ((size_t)(b * TT + c * 128 + k4 * 4)) * DD + h * HDH + d;
#pragma unroll
        for (int jj = 0; jj < 4; ++jj) f[u][jj] = vp[(size_t)jj * DD];
      }
#pragma unroll
      for (int u = 0; u < 4; ++u) {
        int li = (gq * 4 + u) * 128 + tid;
        int d = li & 63;
        int k4 = li >> 6;
        *(v4h*)&kv[d * 136 + k4 * 4] = cvt4(f[u][0], f[u][1], f[u][2], f[u][3]);
      }
    }
    __syncthreads();

    for (int s4 = 0; s4 < 4; ++s4) {
      // A fragment: probs f32 -> f16, float2 reads => ds_load_b64 + v_cvt_pk
      const float2* spa = (const float2*)&ss[l16 * 512 + c * 128 + s4 * 32 + khalf * 8];
      const float2* spb = (const float2*)&ss[l16 * 512 + c * 128 + s4 * 32 + 16 + khalf * 8];
      v16h af;
#pragma unroll
      for (int jj = 0; jj < 4; ++jj) {
        float2 fa = spa[jj];
        af[2 * jj] = (_Float16)fa.x;
        af[2 * jj + 1] = (_Float16)fa.y;
      }
#pragma unroll
      for (int jj = 0; jj < 4; ++jj) {
        float2 fb = spb[jj];
        af[8 + 2 * jj] = (_Float16)fb.x;
        af[8 + 2 * jj + 1] = (_Float16)fb.y;
      }
      const _Float16* bp = &kv[(n0 + l16) * 136 + s4 * 32];
      v16h bf = cat8(*(const v8h*)(bp + khalf * 8), *(const v8h*)(bp + 16 + khalf * 8));
      oacc = wmma16(af, bf, oacc);
    }
    __syncthreads();
  }
#pragma unroll
  for (int r = 0; r < 8; ++r) {
    int m = khalf * 8 + r;
    int col = n0 + l16;
    out[((size_t)(b * TT + q0 + m)) * DD + h * HDH + col] = oacc[r];
  }
}

// ---------------------------------------------------------------------------
// host side
// ---------------------------------------------------------------------------
extern "C" void kernel_launch(void* const* d_in, const int* in_sizes, int n_in,
                              void* d_out, int out_size, void* d_ws, size_t ws_size,
                              hipStream_t stream) {
  (void)in_sizes; (void)n_in; (void)out_size; (void)ws_size;

  const int* enc_ids = (const int*)d_in[0];
  const int* dec_ids = (const int*)d_in[1];
  const float* emb = (const float*)d_in[2];
  auto F = [&](int i) { return (const float*)d_in[i]; };

  const size_t NTOK = (size_t)BBATCH * TT;   // 8192 rows
  const size_t SZ = NTOK * DD;               // 4,194,304 floats

  float* xe = (float*)d_ws;        // encoder activations / enc_out
  float* xd = xe + SZ;             // decoder activations
  float* qb = xd + SZ;
  float* kb = qb + SZ;
  float* vb = kb + SZ;
  float* ab = vb + SZ;             // attention output (concat heads)
  float* tb = ab + SZ;             // gemm temp
  float* hb = tb + SZ;             // FF hidden (8192 x 2048)
  float* logits = (float*)d_out;

  auto gemm = [&](const float* A, int lda, const float* W, const float* bias,
                  const float* resid, float* C, int M, int N, int K, int relu) {
    dim3 g(N / 64, (M + 63) / 64);
    gemm_kernel<<<g, 128, 0, stream>>>(A, lda, W, bias, resid, C, M, N, K, relu);
  };
  auto attn = [&](const float* qp, const float* kp, const float* vp, float* op,
                  const int* iq, const int* ik, int causal) {
    dim3 g(TT / 16, HH, BBATCH);
    attn_kernel<<<g, 128, ATTN_SMEM, stream>>>(qp, kp, vp, op, iq, ik, causal);
  };
  auto ln = [&](const float* in, const float* g, const float* b, float* o) {
    ln_kernel<<<(int)NTOK, 128, 0, stream>>>(in, g, b, o);
  };

  // ---- encoder ----
  embed_kernel<<<(int)NTOK, 128, 0, stream>>>(enc_ids, emb, xe);
  for (int l = 0; l < 6; ++l) {
    int p = 3 + l * 16;
    gemm(xe, DD, F(p + 0), F(p + 1), nullptr, qb, (int)NTOK, DD, DD, 0); // wq,bq
    gemm(xe, DD, F(p + 2), F(p + 3), nullptr, kb, (int)NTOK, DD, DD, 0); // wk,bk
    gemm(xe, DD, F(p + 4), F(p + 5), nullptr, vb, (int)NTOK, DD, DD, 0); // wv,bv
    attn(qb, kb, vb, ab, enc_ids, enc_ids, 0);
    gemm(ab, DD, F(p + 6), F(p + 7), xe, tb, (int)NTOK, DD, DD, 0);      // wo,bo + residual
    ln(tb, F(p + 8), F(p + 9), xe);                                       // ln1
    gemm(xe, DD, F(p + 10), F(p + 11), nullptr, hb, (int)NTOK, 4 * DD, DD, 1); // w1,b1,relu
    gemm(hb, 4 * DD, F(p + 12), F(p + 13), xe, tb, (int)NTOK, DD, 4 * DD, 0);  // w2,b2 + residual
    ln(tb, F(p + 14), F(p + 15), xe);                                     // ln2
  }
  // xe now holds enc_out

  // ---- decoder ----
  embed_kernel<<<(int)NTOK, 128, 0, stream>>>(dec_ids, emb, xd);
  for (int l = 0; l < 6; ++l) {
    int p = 99 + l * 26;
    // masked self-attention
    gemm(xd, DD, F(p + 0), F(p + 1), nullptr, qb, (int)NTOK, DD, DD, 0);
    gemm(xd, DD, F(p + 2), F(p + 3), nullptr, kb, (int)NTOK, DD, DD, 0);
    gemm(xd, DD, F(p + 4), F(p + 5), nullptr, vb, (int)NTOK, DD, DD, 0);
    attn(qb, kb, vb, ab, dec_ids, dec_ids, 1);
    gemm(ab, DD, F(p + 6), F(p + 7), xd, tb, (int)NTOK, DD, DD, 0);
    ln(tb, F(p + 8), F(p + 9), xd);                                       // ln1
    // cross-attention (K,V from enc_out)
    gemm(xd, DD, F(p + 10), F(p + 11), nullptr, qb, (int)NTOK, DD, DD, 0);
    gemm(xe, DD, F(p + 12), F(p + 13), nullptr, kb, (int)NTOK, DD, DD, 0);
    gemm(xe, DD, F(p + 14), F(p + 15), nullptr, vb, (int)NTOK, DD, DD, 0);
    attn(qb, kb, vb, ab, dec_ids, enc_ids, 0);
    gemm(ab, DD, F(p + 16), F(p + 17), xd, tb, (int)NTOK, DD, DD, 0);
    ln(tb, F(p + 18), F(p + 19), xd);                                      // ln2
    // feed-forward
    gemm(xd, DD, F(p + 20), F(p + 21), nullptr, hb, (int)NTOK, 4 * DD, DD, 1);
    gemm(hb, 4 * DD, F(p + 22), F(p + 23), xd, tb, (int)NTOK, DD, 4 * DD, 0);
    ln(tb, F(p + 24), F(p + 25), xd);                                      // ln3
  }

  // ---- final logits: last position only ----
  const float* wout = F(255);
  const float* bout = F(256);
  gemm(xd + (size_t)(TT - 1) * DD, TT * DD, wout, bout, nullptr, logits,
       BBATCH, VV, DD, 0);
}
